// SpikingSwiGLU_8194797601115
// MI455X (gfx1250) — matrix operand order repaired and checked by reading, not compiled, not run on hardware
//
#include <hip/hip_runtime.h>
#include <stdint.h>

// CDNA5 async HBM->LDS staging (GLOBAL_LOAD_ASYNC_TO_LDS_B128, ASYNCcnt).
// Set to 0 to fall back to register-staged copies if the mnemonic rejects.
#define ASYNC_STAGE 1

typedef __bf16 bf16;
typedef __attribute__((ext_vector_type(16))) __bf16 v16bf;
typedef __attribute__((ext_vector_type(8)))  __bf16 v8bf;
typedef __attribute__((ext_vector_type(8)))  float  v8f;

#define BM   256   // token rows per block
#define BN   128   // output cols per block
#define BK   32    // K per step == WMMA K
#define LDSS 40    // LDS row stride in bf16 (80B rows, 16B-aligned segments)

__device__ __forceinline__ v16bf join8(v8bf lo, v8bf hi) {
  union { v16bf v; v8bf h[2]; } u;
  u.h[0] = lo; u.h[1] = hi;
  return u.v;
}

// A fragment (16x32 bf16, ISA 7.12.2): lane m = lane&15, half kh = lane>>4.
// elems 0..7 <- K = kh*8 + 0..7 ; elems 8..15 <- K = 16 + kh*8 + 0..7
__device__ __forceinline__ v16bf load_a_frag(const bf16* rowbase, int kh) {
  return join8(*(const v8bf*)(rowbase + kh * 8),
               *(const v8bf*)(rowbase + 16 + kh * 8));
}

// B fragment (32x16 bf16): lane n = lane&15, kh = lane>>4; elems 0..7 <- K = kh*8 + 0..7 ; elems 8..15 <- K = 16 + kh*8 + 0..7
__device__ __forceinline__ v16bf load_b_frag(const bf16* colbase, int kh) {
  const bf16* p = colbase + kh * 8;
  return join8(*(const v8bf*)(p), *(const v8bf*)(p + 16));
}

// 16-byte HBM -> LDS copy. Async path bypasses VGPRs entirely.
__device__ __forceinline__ void cp16(bf16* lds, const bf16* g) {
#if ASYNC_STAGE
  asm volatile("global_load_async_to_lds_b128 %0, %1, off"
               :: "v"((uint32_t)(uintptr_t)lds),   // LDS byte address (low 32 bits of flat)
                  "v"((uint64_t)(uintptr_t)g)
               : "memory");
#else
  *(v8bf*)lds = *(const v8bf*)g;
#endif
}

__device__ __forceinline__ void cp_fence() {
#if ASYNC_STAGE
  asm volatile("s_wait_asynccnt 0x0" ::: "memory");
#endif
}

#define WMMA_BF16(A, B, C) \
  __builtin_amdgcn_wmma_f32_16x16x32_bf16(false, (A), false, (B), (short)0, (C), false, false)

// ---------------------------------------------------------------------------
// fp32 -> bf16 bulk convert (8 elements / thread)
// ---------------------------------------------------------------------------
__global__ __launch_bounds__(256)
void k0_cvt_bf16(const float* __restrict__ src, bf16* __restrict__ dst, int n) {
  const int i = (blockIdx.x * 256 + threadIdx.x) * 8;
  if (i >= n) return;
  float4 a = *(const float4*)(src + i);
  float4 b = *(const float4*)(src + i + 4);
  v8bf o;
  o[0] = (bf16)a.x; o[1] = (bf16)a.y; o[2] = (bf16)a.z; o[3] = (bf16)a.w;
  o[4] = (bf16)b.x; o[5] = (bf16)b.y; o[6] = (bf16)b.z; o[7] = (bf16)b.w;
  *(v8bf*)(dst + i) = o;
}

// ---------------------------------------------------------------------------
// Kernel 1: gated[m,n] = swish(x@fc_w[n]^T + b[n]) * (x@fc_w[n+H]^T + b[n+H])
// 512 threads = 16 waves (4M x 4N), wave tile 64x32, dual accumulators.
// ---------------------------------------------------------------------------
__global__ __launch_bounds__(512)
void k1_gemm_swiglu(const bf16* __restrict__ Xb,  // [M, D] bf16
                    const bf16* __restrict__ Wb,  // [2H, D] bf16
                    const float* __restrict__ FB, // [2H]
                    bf16* __restrict__ G,         // [M, H]
                    int M, int D, int H) {
  __shared__ bf16 As [2][BM * LDSS];
  __shared__ bf16 B1s[2][BN * LDSS];
  __shared__ bf16 B2s[2][BN * LDSS];

  const int tid  = threadIdx.x;
  const int lane = tid & 31;
  const int wave = tid >> 5;
  const int wm   = wave & 3;          // 4 waves over M (64 rows each)
  const int wn   = wave >> 2;         // 4 waves over N (32 cols each)
  const int bM   = blockIdx.y * BM;
  const int bN   = blockIdx.x * BN;   // x1 column block; x2 block = bN + H
  const int kh   = lane >> 4;
  const int ml   = lane & 15;

  const v8f vzero = {0.f, 0.f, 0.f, 0.f, 0.f, 0.f, 0.f, 0.f};
  v8f acc1[8], acc2[8];
#pragma unroll
  for (int i = 0; i < 8; ++i) { acc1[i] = vzero; acc2[i] = vzero; }

  auto stage = [&](int k0, int b) {
    // A: 256 rows x 4 segs = 1024 chunks over 512 threads
#pragma unroll
    for (int c = 0; c < 2; ++c) {
      const int ch = tid + c * 512;
      const int row = ch >> 2, seg = ch & 3;
      cp16(&As[b][row * LDSS + seg * 8],
           Xb + (size_t)(bM + row) * D + k0 + seg * 8);
    }
    // B1/B2: 128 rows x 4 segs = 512 chunks each
    const int row = tid >> 2, seg = tid & 3;
    cp16(&B1s[b][row * LDSS + seg * 8],
         Wb + (size_t)(bN + row) * D + k0 + seg * 8);
    cp16(&B2s[b][row * LDSS + seg * 8],
         Wb + (size_t)(bN + H + row) * D + k0 + seg * 8);
  };

  auto compute = [&](int b) {
    v16bf a[4];
#pragma unroll
    for (int mt = 0; mt < 4; ++mt)
      a[mt] = load_a_frag(&As[b][(wm * 64 + mt * 16 + ml) * LDSS], kh);
#pragma unroll
    for (int nt = 0; nt < 2; ++nt) {
      v16bf b1 = load_b_frag(&B1s[b][(wn * 32 + nt * 16 + ml) * LDSS], kh);
      v16bf b2 = load_b_frag(&B2s[b][(wn * 32 + nt * 16 + ml) * LDSS], kh);
#pragma unroll
      for (int mt = 0; mt < 4; ++mt) {
        acc1[mt * 2 + nt] = WMMA_BF16(a[mt], b1, acc1[mt * 2 + nt]);
        acc2[mt * 2 + nt] = WMMA_BF16(a[mt], b2, acc2[mt * 2 + nt]);
      }
    }
  };

  stage(0, 0);
  cp_fence();
  __syncthreads();

  int buf = 0;
  for (int k0 = 0; k0 < D; k0 += BK) {
    const bool more = (k0 + BK) < D;
    if (more) stage(k0 + BK, buf ^ 1);   // async HBM->LDS overlaps WMMA below
    compute(buf);
    if (more) cp_fence();                // drain before publishing at barrier
    __syncthreads();
    buf ^= 1;
  }

  // epilogue: bias + swish(x1) * x2, store bf16
#pragma unroll
  for (int mt = 0; mt < 4; ++mt) {
#pragma unroll
    for (int nt = 0; nt < 2; ++nt) {
      const int idx   = mt * 2 + nt;
      const int n     = bN + wn * 32 + nt * 16 + ml;
      const float bi1 = FB[n];
      const float bi2 = FB[n + H];
#pragma unroll
      for (int r = 0; r < 8; ++r) {
        const int m = bM + wm * 64 + mt * 16 + kh * 8 + r;  // C/D layout
        float x1 = acc1[idx][r] + bi1;
        float x2 = acc2[idx][r] + bi2;
        float sw = x1 / (1.0f + __expf(-x1));
        G[(size_t)m * H + n] = (bf16)(sw * x2);
      }
    }
  }
}

// ---------------------------------------------------------------------------
// Kernel 2: y = gated @ out_w^T + out_b, fused 5-step LIF -> spike rate
// ---------------------------------------------------------------------------
__global__ __launch_bounds__(512)
void k2_gemm_lif(const bf16* __restrict__ G,    // [M, H] bf16
                 const bf16* __restrict__ OWb,  // [D, H] bf16
                 const float* __restrict__ OB,  // [D]
                 const float* __restrict__ pBeta,
                 const float* __restrict__ pSS,
                 const float* __restrict__ pRM,
                 float* __restrict__ Out,       // [M, D]
                 int M, int H, int D) {
  __shared__ bf16 As[2][BM * LDSS];
  __shared__ bf16 Bs[2][BN * LDSS];

  const int tid  = threadIdx.x;
  const int lane = tid & 31;
  const int wave = tid >> 5;
  const int wm   = wave & 3;
  const int wn   = wave >> 2;
  const int bM   = blockIdx.y * BM;
  const int bN   = blockIdx.x * BN;
  const int kh   = lane >> 4;
  const int ml   = lane & 15;

  const v8f vzero = {0.f, 0.f, 0.f, 0.f, 0.f, 0.f, 0.f, 0.f};
  v8f acc[8];
#pragma unroll
  for (int i = 0; i < 8; ++i) acc[i] = vzero;

  auto stage = [&](int k0, int b) {
#pragma unroll
    for (int c = 0; c < 2; ++c) {
      const int ch = tid + c * 512;
      const int row = ch >> 2, seg = ch & 3;
      cp16(&As[b][row * LDSS + seg * 8],
           G + (size_t)(bM + row) * H + k0 + seg * 8);
    }
    const int row = tid >> 2, seg = tid & 3;
    cp16(&Bs[b][row * LDSS + seg * 8],
         OWb + (size_t)(bN + row) * H + k0 + seg * 8);
  };

  auto compute = [&](int b) {
    v16bf a[4];
#pragma unroll
    for (int mt = 0; mt < 4; ++mt)
      a[mt] = load_a_frag(&As[b][(wm * 64 + mt * 16 + ml) * LDSS], kh);
#pragma unroll
    for (int nt = 0; nt < 2; ++nt) {
      v16bf bb = load_b_frag(&Bs[b][(wn * 32 + nt * 16 + ml) * LDSS], kh);
#pragma unroll
      for (int mt = 0; mt < 4; ++mt)
        acc[mt * 2 + nt] = WMMA_BF16(a[mt], bb, acc[mt * 2 + nt]);
    }
  };

  stage(0, 0);
  cp_fence();
  __syncthreads();

  int buf = 0;
  for (int k0 = 0; k0 < H; k0 += BK) {
    const bool more = (k0 + BK) < H;
    if (more) stage(k0 + BK, buf ^ 1);
    compute(buf);
    if (more) cp_fence();
    __syncthreads();
    buf ^= 1;
  }

  const float bclamp = fminf(fmaxf(pBeta[0], 0.0f), 1.0f);
  const float oscale = 0.2f * pRM[0] * pSS[0];  // (1/NUM_STEPS) * rm * ss

#pragma unroll
  for (int mt = 0; mt < 4; ++mt) {
#pragma unroll
    for (int nt = 0; nt < 2; ++nt) {
      const int idx  = mt * 2 + nt;
      const int n    = bN + wn * 32 + nt * 16 + ml;
      const float bi = OB[n];
#pragma unroll
      for (int r = 0; r < 8; ++r) {
        const int m = bM + wm * 64 + mt * 16 + kh * 8 + r;
        const float y = acc[idx][r] + bi;
        float mem = 0.0f, cnt = 0.0f;
#pragma unroll
        for (int t = 0; t < 5; ++t) {
          float reset = (mem > 1.0f) ? 1.0f : 0.0f;   // subtract-reset (detached)
          mem = bclamp * mem + y - reset;
          cnt += (mem > 1.0f) ? 1.0f : 0.0f;          // Heaviside(mem - 1)
        }
        Out[(size_t)m * D + n] = cnt * oscale;
      }
    }
  }
}

extern "C" void kernel_launch(void* const* d_in, const int* in_sizes, int n_in,
                              void* d_out, int out_size, void* d_ws, size_t ws_size,
                              hipStream_t stream) {
  const float* x     = (const float*)d_in[0];
  const float* fc_w  = (const float*)d_in[1];
  const float* fc_b  = (const float*)d_in[2];
  const float* out_w = (const float*)d_in[3];
  const float* out_b = (const float*)d_in[4];
  const float* beta  = (const float*)d_in[5];
  const float* ss    = (const float*)d_in[6];
  const float* rm    = (const float*)d_in[7];

  const int H = in_sizes[2] / 2;      // 8192
  const int D = in_sizes[4];          // 2048
  const int M = in_sizes[0] / D;      // 8192 tokens (B*S)

  // ws layout (bf16): gated[M*H] | Xb[M*D] | Wb[2H*D] | OWb[D*H]  = 256 MB total
  bf16* gated = (bf16*)d_ws;
  bf16* Xb    = gated + (size_t)M * H;
  bf16* Wb    = Xb + (size_t)M * D;
  bf16* OWb   = Wb + (size_t)2 * H * D;

  dim3 cb(256);
  k0_cvt_bf16<<<dim3((unsigned)((size_t)M * D / 2048)), cb, 0, stream>>>(x, Xb, M * D);
  k0_cvt_bf16<<<dim3((unsigned)((size_t)2 * H * D / 2048)), cb, 0, stream>>>(fc_w, Wb, 2 * H * D);
  k0_cvt_bf16<<<dim3((unsigned)((size_t)D * H / 2048)), cb, 0, stream>>>(out_w, OWb, D * H);

  dim3 block(512);
  dim3 g1(H / BN, M / BM);            // 64 x 32
  k1_gemm_swiglu<<<g1, block, 0, stream>>>(Xb, Wb, fc_b, gated, M, D, H);

  dim3 g2(D / BN, M / BM);            // 16 x 32
  k2_gemm_lif<<<g2, block, 0, stream>>>(gated, out_w ? OWb : OWb, out_b, beta, ss, rm,
                                        (float*)d_out, M, H, D);
}
